// Compositional_Self_Attention_25451976196684
// MI455X (gfx1250) — matrix operand order repaired and checked by reading, not compiled, not run on hardware
//
#include <hip/hip_runtime.h>

// ---------------------------------------------------------------------------
// Compositional self-attention block for MI455X (gfx1250, wave32, WMMA).
// - All GEMM math on v_wmma_f32_16x16x32_f16 (f32 accumulation).
// - Weights stored transposed (N-major) in f16 so every WMMA fragment load
//   from LDS is K-contiguous (ds_load_b128), no scalar u16 gather.
// - GEMM tiles double-buffered and staged with CDNA5 async paths:
//   GLOBAL_LOAD_ASYNC_TO_LDS_B128 (ASYNCcnt) for A, Tensor Data Mover
//   (tensor_load_to_lds, TENSORcnt) for B. Fallback: sync copies.
// - Attention is flash-style (no 268MB score tensor), fused with the rule
//   composition; Wqv/bqv/bs eliminated (softmax shift invariance in r).
// ---------------------------------------------------------------------------

typedef _Float16 half_t;
typedef __attribute__((ext_vector_type(16))) _Float16 v16h;
typedef __attribute__((ext_vector_type(8)))  float    v8f;
typedef unsigned int uint32x4 __attribute__((ext_vector_type(4)));
typedef int          int32x4  __attribute__((ext_vector_type(4)));
typedef int          int32x8  __attribute__((ext_vector_type(8)));

#define XSTR(a) STR(a)
#define STR(a) #a
#pragma message("probe: clang major = " XSTR(__clang_major__))

#if __has_builtin(__builtin_amdgcn_global_load_async_to_lds_b128) && \
    __has_builtin(__builtin_amdgcn_s_wait_asynccnt)
#define HAVE_ASYNC 1
#else
#define HAVE_ASYNC 0
#endif

#if __has_builtin(__builtin_amdgcn_tensor_load_to_lds) && \
    __has_builtin(__builtin_amdgcn_s_wait_tensorcnt)
#define HAVE_TDM 1
#else
#define HAVE_TDM 0
#endif

#define DIM   1024
#define SEQ   1024
#define BATCH 4
#define NH    16
#define HD    64
#define NR    8
#define NV    (NR * HD)      // 512
#define NTOK  (BATCH * SEQ)  // 4096

__device__ inline v8f vzero() {
  v8f z;
#pragma unroll
  for (int i = 0; i < 8; ++i) z[i] = 0.f;
  return z;
}

__device__ inline v8f wmma16(v16h a, v16h b, v8f c) {
  return __builtin_amdgcn_wmma_f32_16x16x32_f16(false, a, false, b, (short)0, c,
                                                false, false);
}

union FragU {
  v16h v;
  unsigned u[8];
  half_t h[16];
};

// A fragment (16x32 f16, contiguous k order, not ISA table order): p = rowBase(lane&15)*ld + 16*(lane>=16).
// elements 0..15 <-> K=kb+0..15, kb=16*(lane>=16)
__device__ inline v16h load_afrag(const half_t* p) {
  FragU f;
#pragma unroll
  for (int j = 0; j < 4; ++j) f.u[j] = *(const unsigned*)(p + 2 * j);
#pragma unroll
  for (int j = 0; j < 4; ++j) f.u[4 + j] = *(const unsigned*)(p + 8 + 2 * j);
  return f.v;
}

// B fragment (32x16): lane holds column (lane&15), K = 16*(lane>=16)+i.
// p positioned at this lane's first K element; 16 K-values contiguous.
__device__ inline v16h load_bfrag_contig(const half_t* p) {
  FragU f;
#pragma unroll
  for (int j = 0; j < 8; ++j) f.u[j] = *(const unsigned*)(p + 2 * j);
  return f.v;
}

#if HAVE_ASYNC
// Builtin signature (from clang-23 diagnostic): param1 = int4(vector_size 16)
// in AS1 ("__device__"), param2 = AS3, then imm offset, imm cpol.
typedef int v4i_t __attribute__((vector_size(16)));
typedef __attribute__((address_space(1))) v4i_t gv4i_t;
typedef __attribute__((address_space(3))) v4i_t lv4i_t;
__device__ inline void async_b128(half_t* ldsDst, const half_t* gSrc) {
  __builtin_amdgcn_global_load_async_to_lds_b128(
      (gv4i_t*)gSrc, (lv4i_t*)ldsDst, 0, 0);
}
#endif

#if HAVE_TDM
typedef __attribute__((address_space(3))) char lds_char_t;
__device__ inline unsigned lds_off(void* p) {
  return (unsigned)(unsigned long long)(lds_char_t*)p;
}
#endif

// ---------------------------------------------------------------------------
// f32 -> f16 transposing weight conversion: dst[n*K + k] = (f16)src[k*N + n]
// ---------------------------------------------------------------------------
__global__ void __launch_bounds__(256)
cvt_t_kernel(const float* __restrict__ src, half_t* __restrict__ dst,
             int K, int N) {
  int i = blockIdx.x * 256 + threadIdx.x;
  if (i < K * N) {
    int k = i / N, n = i - k * N;
    dst[(size_t)n * K + k] = (half_t)src[i];
  }
}

// ---------------------------------------------------------------------------
// LayerNorm over DIM=1024, one block (8 waves) per row, f16 output
// ---------------------------------------------------------------------------
__global__ void __launch_bounds__(256)
ln_kernel(const float* __restrict__ x, const float* __restrict__ g,
          const float* __restrict__ be, half_t* __restrict__ out) {
  __shared__ float wsum[8], wsq[8];
  const int row = blockIdx.x;
  const int tid = threadIdx.x;
  float4 v = ((const float4*)(x + (size_t)row * DIM))[tid];
  float s = v.x + v.y + v.z + v.w;
  float q = v.x * v.x + v.y * v.y + v.z * v.z + v.w * v.w;
#pragma unroll
  for (int m = 1; m < 32; m <<= 1) { s += __shfl_xor(s, m); q += __shfl_xor(q, m); }
  if ((tid & 31) == 0) { wsum[tid >> 5] = s; wsq[tid >> 5] = q; }
  __syncthreads();
  float ts = 0.f, tq = 0.f;
#pragma unroll
  for (int i = 0; i < 8; ++i) { ts += wsum[i]; tq += wsq[i]; }
  const float mean = ts * (1.f / DIM);
  const float var = tq * (1.f / DIM) - mean * mean;
  const float rstd = rsqrtf(var + 1e-5f);
  float4 gg = ((const float4*)g)[tid];
  float4 bb = ((const float4*)be)[tid];
  union { half_t h[4]; float2 f2; } o;
  o.h[0] = (half_t)((v.x - mean) * rstd * gg.x + bb.x);
  o.h[1] = (half_t)((v.y - mean) * rstd * gg.y + bb.y);
  o.h[2] = (half_t)((v.z - mean) * rstd * gg.z + bb.z);
  o.h[3] = (half_t)((v.w - mean) * rstd * gg.w + bb.w);
  *(float2*)(out + (size_t)row * DIM + tid * 4) = o.f2;
}

// ---------------------------------------------------------------------------
// WMMA GEMM: C[M,N] = A[M,K](f16 row-major) * BT[N,K](f16 N-major)
// (+bias, *alpha, relu, +f32 residual; f32 and/or f16 (opt. transposed) out).
// 256 thr = 8 waves; tile 64x128, BK=32; both LDS tiles K-contiguous (ld=40);
// double-buffered; A staged via async-to-LDS, B via TDM when available.
// ---------------------------------------------------------------------------
#define BM 64
#define BN 128
#define BK 32
#define LDT 40  // 32 + 8 pad (TDM pad: interval 16 dwords, amount 4 dwords)

#if HAVE_TDM
#define APT 1   // async ops per thread per tile (A only)
#else
#define APT 3   // A(1) + B(2)
#endif

__global__ void __launch_bounds__(256)
gemm_kernel(const half_t* __restrict__ A, const half_t* __restrict__ BT,
            const float* __restrict__ bias, const float* __restrict__ res,
            float* __restrict__ Cf, half_t* __restrict__ Ch,
            int M, int N, int K, float alpha, int relu, int tpose) {
  __shared__ half_t As[2][BM * LDT];
  __shared__ half_t Bs[2][BN * LDT];
  const int tid = threadIdx.x;
  const int lane = tid & 31, w = tid >> 5;
  const int ln = lane & 15, hi = lane >> 4;
  const int wm = (w >> 2) * 32, wn = (w & 3) * 32;
  const int rowBase = blockIdx.y * BM, colBase = blockIdx.x * BN;

  v8f acc00 = vzero(), acc01 = vzero(), acc10 = vzero(), acc11 = vzero();

  // A tile: 64 rows x 32 k -> 256 chunks of 8 f16, 1 per thread
  const int arow = tid >> 2, ach = (tid & 3) * 8;
  // B tile: 128 rows(n) x 32 k -> 512 chunks, 2 per thread
  const int bn0 = tid >> 2, bc0 = (tid & 3) * 8;                  // rows 0..63
  const int bn1 = (tid + 256) >> 2, bc1 = ((tid + 256) & 3) * 8;  // 64..127

  auto issue_tile = [&](int kk, int bufIdx) {
    half_t* Asb = &As[bufIdx][0];
    half_t* Bsb = &Bs[bufIdx][0];
    const half_t* aSrc = A + (size_t)(rowBase + arow) * K + kk + ach;
#if HAVE_ASYNC
    async_b128(Asb + arow * LDT + ach, aSrc);
#else
    *(float4*)(Asb + arow * LDT + ach) = *(const float4*)aSrc;
#endif
#if HAVE_TDM
    if (tid < 32) {  // one wave issues the TDM descriptor for the B tile
      const unsigned long long ga =
          (unsigned long long)(BT + (size_t)colBase * K + kk);
      uint32x4 g0 = {1u,                       // count=1, user descriptor
                     lds_off(Bsb),             // lds_addr
                     (unsigned)ga,             // global_addr[31:0]
                     (unsigned)((ga >> 32) & 0x01FFFFFFull) | (2u << 30)};
      int32x8 g1 = {(1 << 16) | (1 << 20) | (3 << 22) | (3 << 25),
                    // data_size=2B, pad_enable, interval=16dw, amount=4dw
                    (BK << 16),   // tensor_dim0 = 32 (low 16 in [31:16])
                    (BN << 16),   // dim0 hi=0 | tensor_dim1 = 128
                    (BK << 16),   // dim1 hi=0 | tile_dim0 = 32
                    BN,           // tile_dim1 = 128, tile_dim2 = 0
                    K,            // tensor_dim0_stride = K elements
                    0, 0};
      int32x4 gz = {0, 0, 0, 0};
#if __clang_major__ >= 23
      int32x8 gz8 = {0, 0, 0, 0, 0, 0, 0, 0};
      __builtin_amdgcn_tensor_load_to_lds(g0, g1, gz, gz, gz8, 0);
#else
      __builtin_amdgcn_tensor_load_to_lds(g0, g1, gz, gz, 0);
#endif
    }
#elif HAVE_ASYNC
    async_b128(Bsb + bn0 * LDT + bc0, BT + (size_t)(colBase + bn0) * K + kk + bc0);
    async_b128(Bsb + bn1 * LDT + bc1, BT + (size_t)(colBase + bn1) * K + kk + bc1);
#else
    *(float4*)(Bsb + bn0 * LDT + bc0) =
        *(const float4*)(BT + (size_t)(colBase + bn0) * K + kk + bc0);
    *(float4*)(Bsb + bn1 * LDT + bc1) =
        *(const float4*)(BT + (size_t)(colBase + bn1) * K + kk + bc1);
#endif
  };

  issue_tile(0, 0);
  int buf = 0;
  for (int k0 = 0; k0 < K; k0 += BK) {
    const bool more = (k0 + BK) < K;
    if (more) issue_tile(k0 + BK, buf ^ 1);
    if (more) {
#if HAVE_ASYNC
      __builtin_amdgcn_s_wait_asynccnt(APT);
#endif
#if HAVE_TDM
      if (tid < 32) __builtin_amdgcn_s_wait_tensorcnt(1);
#endif
    } else {
#if HAVE_ASYNC
      __builtin_amdgcn_s_wait_asynccnt(0);
#endif
#if HAVE_TDM
      if (tid < 32) __builtin_amdgcn_s_wait_tensorcnt(0);
#endif
    }
    __syncthreads();

    const half_t* Asb = &As[buf][0];
    const half_t* Bsb = &Bs[buf][0];
    v16h a0 = load_afrag(Asb + (wm + ln) * LDT + 16 * hi);
    v16h a1 = load_afrag(Asb + (wm + 16 + ln) * LDT + 16 * hi);
    v16h b0 = load_bfrag_contig(Bsb + (wn + ln) * LDT + 16 * hi);
    v16h b1 = load_bfrag_contig(Bsb + (wn + 16 + ln) * LDT + 16 * hi);
    acc00 = wmma16(a0, b0, acc00);
    acc01 = wmma16(a0, b1, acc01);
    acc10 = wmma16(a1, b0, acc10);
    acc11 = wmma16(a1, b1, acc11);
    __syncthreads();
    buf ^= 1;
  }

  auto store_tile = [&](v8f a, int ti, int tj) {
    const int col = colBase + wn + tj * 16 + ln;
    const float bv = bias ? bias[col] : 0.f;
#pragma unroll
    for (int j = 0; j < 8; ++j) {
      const int row = rowBase + wm + ti * 16 + (hi << 3) + j;
      float v = (a[j] + bv) * alpha;
      if (relu) v = fmaxf(v, 0.f);
      if (res) v += res[(size_t)row * N + col];
      if (Cf) Cf[(size_t)row * N + col] = v;
      if (Ch) {
        if (tpose)  // per-batch transpose: dst[b][col][s]
          Ch[((size_t)(row >> 10) * N + col) * SEQ + (row & (SEQ - 1))] = (half_t)v;
        else
          Ch[(size_t)row * N + col] = (half_t)v;
      }
    }
  };
  store_tile(acc00, 0, 0);
  store_tile(acc01, 0, 1);
  store_tile(acc10, 1, 0);
  store_tile(acc11, 1, 1);
}

// ---------------------------------------------------------------------------
// Fused flash attention + rule composition.
// ---------------------------------------------------------------------------
__global__ void __launch_bounds__(256)
attn_kernel(const half_t* __restrict__ Q, const half_t* __restrict__ Km,
            const half_t* __restrict__ Vt,  // [BATCH][NV][SEQ]
            const float* __restrict__ Ws,   // 96 floats; uses Ws[32..95]
            half_t* __restrict__ Out)       // [NTOK][DIM]
{
  __shared__ half_t Pl[2][16 * 32];
  __shared__ float Lg[2][16][8];
  __shared__ float Acc[2][16][64];

  const int qb = blockIdx.x, h = blockIdx.y, b = blockIdx.z;
  const int tid = threadIdx.x, lane = tid & 31, w = tid >> 5;
  const int wq = w >> 2, wc = w & 3;
  const int ln = lane & 15, hi = lane >> 4;

  for (int i = tid; i < 2 * 16 * 64; i += 256) ((float*)Acc)[i] = 0.f;

  const int qrow0 = qb * 32 + wq * 16;
  const half_t* qptr = Q + ((size_t)(b * SEQ + qrow0 + ln)) * DIM + h * HD + 16 * hi;
  const v16h qa0 = load_afrag(qptr);       // 1/sqrt(HD) folded into Q
  const v16h qa1 = load_afrag(qptr + 32);

  v8f acc[8];
#pragma unroll
  for (int t = 0; t < 8; ++t) acc[t] = vzero();
  float mrow[8], lrow[8];
#pragma unroll
  for (int j = 0; j < 8; ++j) { mrow[j] = -1e30f; lrow[j] = 0.f; }

  const int cbase = wc * 128;
  const half_t* vcol = Vt + ((size_t)(b * NV + cbase + ln)) * SEQ + 16 * hi;

  for (int k0 = 0; k0 < SEQ; k0 += 32) {
    const half_t* krow = Km + ((size_t)(b * SEQ + k0 + ln)) * DIM + h * HD + 16 * hi;
    v16h kb00 = load_bfrag_contig(krow);
    v16h kb01 = load_bfrag_contig(krow + 32);
    v16h kb10 = load_bfrag_contig(krow + (size_t)16 * DIM);
    v16h kb11 = load_bfrag_contig(krow + (size_t)16 * DIM + 32);
    v8f s_lo = wmma16(qa0, kb00, vzero());
    s_lo = wmma16(qa1, kb01, s_lo);
    v8f s_hi = wmma16(qa0, kb10, vzero());
    s_hi = wmma16(qa1, kb11, s_hi);

    float mnew[8], corr[8], p_lo[8], p_hi[8];
#pragma unroll
    for (int j = 0; j < 8; ++j) {
      float m = fmaxf(s_lo[j], s_hi[j]);
      m = fmaxf(m, __shfl_xor(m, 1));
      m = fmaxf(m, __shfl_xor(m, 2));
      m = fmaxf(m, __shfl_xor(m, 4));
      m = fmaxf(m, __shfl_xor(m, 8));
      mnew[j] = fmaxf(mrow[j], m);
      corr[j] = __expf(mrow[j] - mnew[j]);
      mrow[j] = mnew[j];
      p_lo[j] = __expf(s_lo[j] - mnew[j]);
      p_hi[j] = __expf(s_hi[j] - mnew[j]);
      float rs = p_lo[j] + p_hi[j];
      rs += __shfl_xor(rs, 1);
      rs += __shfl_xor(rs, 2);
      rs += __shfl_xor(rs, 4);
      rs += __shfl_xor(rs, 8);
      lrow[j] = lrow[j] * corr[j] + rs;
    }
#pragma unroll
    for (int t = 0; t < 8; ++t)
#pragma unroll
      for (int j = 0; j < 8; ++j) acc[t][j] *= corr[j];

    if (wc == 0) {
#pragma unroll
      for (int j = 0; j < 8; ++j) {
        const int row = (hi << 3) + j;
        Pl[wq][row * 32 + ln] = (half_t)p_lo[j];
        Pl[wq][row * 32 + 16 + ln] = (half_t)p_hi[j];
      }
    }
    __syncthreads();
    const v16h pa = load_afrag(&Pl[wq][ln * 32 + 16 * hi]);
#pragma unroll
    for (int t = 0; t < 8; ++t) {
      v16h vb = load_bfrag_contig(vcol + (size_t)t * 16 * SEQ + k0);
      acc[t] = wmma16(pa, vb, acc[t]);
    }
    __syncthreads();
  }

#pragma unroll
  for (int j = 0; j < 8; ++j) {
    const float inv = 1.f / lrow[j];
#pragma unroll
    for (int t = 0; t < 8; ++t) acc[t][j] *= inv;
  }

  const float* ws2 = Ws + 32;
#pragma unroll
  for (int rr = 0; rr < 2; ++rr) {
#pragma unroll
    for (int j = 0; j < 8; ++j) {
      float s = 0.f;
#pragma unroll
      for (int tl = 0; tl < 4; ++tl) s += acc[rr * 4 + tl][j] * ws2[tl * 16 + ln];
      s += __shfl_xor(s, 1);
      s += __shfl_xor(s, 2);
      s += __shfl_xor(s, 4);
      s += __shfl_xor(s, 8);
      if (ln == 0) Lg[wq][(hi << 3) + j][wc * 2 + rr] = s;
    }
  }
  __syncthreads();

  float w0[8], w1[8];
#pragma unroll
  for (int j = 0; j < 8; ++j) {
    const float* lg = Lg[wq][(hi << 3) + j];
    float mx = lg[0];
#pragma unroll
    for (int r = 1; r < 8; ++r) mx = fmaxf(mx, lg[r]);
    float den = 0.f;
#pragma unroll
    for (int r = 0; r < 8; ++r) den += __expf(lg[r] - mx);
    const float id = 1.f / den;
    w0[j] = __expf(lg[wc * 2] - mx) * id;
    w1[j] = __expf(lg[wc * 2 + 1] - mx) * id;
  }
  for (int turn = 0; turn < 4; ++turn) {  // deterministic rule reduction
    if (wc == turn) {
#pragma unroll
      for (int j = 0; j < 8; ++j) {
        const int row = (hi << 3) + j;
#pragma unroll
        for (int tl = 0; tl < 4; ++tl)
          Acc[wq][row][tl * 16 + ln] += acc[tl][j] * w0[j] + acc[4 + tl][j] * w1[j];
      }
    }
    __syncthreads();
  }

  {
    const int row = tid >> 3;
    const int c0 = (tid & 7) * 8;
    union { half_t hh[8]; float4 f4; } o;
#pragma unroll
    for (int i = 0; i < 8; ++i) o.hh[i] = (half_t)Acc[row >> 4][row & 15][c0 + i];
    *(float4*)(Out + ((size_t)(b * SEQ + qb * 32 + row)) * DIM + h * HD + c0) = o.f4;
  }
}

// ---------------------------------------------------------------------------
// Host orchestration (all on `stream`; workspace ~59 MB, 256B-aligned)
// ---------------------------------------------------------------------------
extern "C" void kernel_launch(void* const* d_in, const int* in_sizes, int n_in,
                              void* d_out, int out_size, void* d_ws, size_t ws_size,
                              hipStream_t stream) {
  (void)in_sizes; (void)n_in; (void)out_size; (void)ws_size;
  const float* x   = (const float*)d_in[0];
  const float* Wq  = (const float*)d_in[1];
  const float* bq  = (const float*)d_in[2];
  const float* Wk  = (const float*)d_in[3];
  const float* bk  = (const float*)d_in[4];
  const float* Wv  = (const float*)d_in[5];
  const float* bv  = (const float*)d_in[6];
  // d_in[7]=Wqv, d_in[8]=bqv, d_in[10]=bs: algebraically eliminated
  const float* Ws  = (const float*)d_in[9];
  const float* Wf  = (const float*)d_in[11];
  const float* bf  = (const float*)d_in[12];
  const float* W1  = (const float*)d_in[13];
  const float* b1  = (const float*)d_in[14];
  const float* W2  = (const float*)d_in[15];
  const float* b2  = (const float*)d_in[16];
  const float* g1  = (const float*)d_in[17];
  const float* be1 = (const float*)d_in[18];
  const float* g2  = (const float*)d_in[19];
  const float* be2 = (const float*)d_in[20];
  float* out = (float*)d_out;

  char* ws = (char*)d_ws;
  size_t off = 0;
  auto alloc = [&](size_t bytes) -> char* {
    char* p = ws + off;
    off += (bytes + 255) & ~(size_t)255;
    return p;
  };
  half_t* WqT = (half_t*)alloc((size_t)DIM * DIM * 2);
  half_t* WkT = (half_t*)alloc((size_t)DIM * DIM * 2);
  half_t* WvT = (half_t*)alloc((size_t)DIM * NV * 2);
  half_t* WfT = (half_t*)alloc((size_t)DIM * DIM * 2);
  half_t* W1T = (half_t*)alloc((size_t)DIM * 2 * DIM * 2);
  half_t* W2T = (half_t*)alloc((size_t)2 * DIM * DIM * 2);
  half_t* xn16 = (half_t*)alloc((size_t)NTOK * DIM * 2);  // } adjacent:
  half_t* at16 = (half_t*)alloc((size_t)NTOK * DIM * 2);  // } reused as h1
  half_t* q16  = (half_t*)alloc((size_t)NTOK * DIM * 2);  // } adjacent:
  half_t* k16  = (half_t*)alloc((size_t)NTOK * DIM * 2);  // } reused as y1
  half_t* vT16 = (half_t*)alloc((size_t)NTOK * NV * 2);
  half_t* on16 = (half_t*)alloc((size_t)NTOK * DIM * 2);
  float*  y1   = (float*)q16;  // alive only after q16/k16 are dead
  half_t* h116 = xn16;         // alive only after xn16/at16 are dead
  (void)k16;

  const dim3 blk(256);
  auto cvt_t = [&](const float* s, half_t* d, int K, int N) {
    cvt_t_kernel<<<dim3((K * N + 255) / 256), blk, 0, stream>>>(s, d, K, N);
  };
  auto gemm = [&](const half_t* A, const half_t* Bt, const float* bias,
                  const float* res, float* Cf, half_t* Ch, int M, int N, int K,
                  float alpha, int relu, int tpose) {
    gemm_kernel<<<dim3(N / BN, M / BM), blk, 0, stream>>>(A, Bt, bias, res, Cf,
                                                          Ch, M, N, K, alpha,
                                                          relu, tpose);
  };

  // 1) weight conversion to transposed f16 (one-time, K-contiguous rows)
  cvt_t(Wq, WqT, DIM, DIM);
  cvt_t(Wk, WkT, DIM, DIM);
  cvt_t(Wv, WvT, DIM, NV);
  cvt_t(Wf, WfT, DIM, DIM);
  cvt_t(W1, W1T, DIM, 2 * DIM);
  cvt_t(W2, W2T, 2 * DIM, DIM);

  // 2) LN1
  ln_kernel<<<dim3(NTOK), blk, 0, stream>>>(x, g1, be1, xn16);

  // 3) projections (1/sqrt(HD) folded into Q; V stored transposed per batch)
  gemm(xn16, WqT, bq, nullptr, nullptr, q16, NTOK, DIM, DIM, 0.125f, 0, 0);
  gemm(xn16, WkT, bk, nullptr, nullptr, k16, NTOK, DIM, DIM, 1.f, 0, 0);
  gemm(xn16, WvT, bv, nullptr, nullptr, vT16, NTOK, NV, DIM, 1.f, 0, 1);

  // 4) fused flash attention + rule composition
  attn_kernel<<<dim3(SEQ / 32, NH, BATCH), blk, 0, stream>>>(q16, k16, vT16,
                                                             Ws, at16);

  // 5) output projection + residual (f32)
  gemm(at16, WfT, bf, x, y1, nullptr, NTOK, DIM, DIM, 1.f, 0, 0);

  // 6) LN2 + FFN + residual
  ln_kernel<<<dim3(NTOK), blk, 0, stream>>>(y1, g2, be2, on16);
  gemm(on16, W1T, b1, nullptr, nullptr, h116, NTOK, 2 * DIM, DIM, 1.f, 1, 0);
  gemm(h116, W2T, b2, y1, out, nullptr, NTOK, DIM, 2 * DIM, 1.f, 0, 0);
}
